// Message_11081015623791
// MI455X (gfx1250) — matrix operand order repaired and checked by reading, not compiled, not run on hardware
//
#include <hip/hip_runtime.h>
#include <hip/hip_bf16.h>

// ---------------------------------------------------------------------------
// Fused GNN edge-message kernel for MI455X (gfx1250, wave32, WMMA).
//   pack:  f32 weights -> f16 transposed weights in d_ws (once)
//   edge:  m = MLP_phi(inv[src]) * MLP_w(PE(dist))   [E,192] via v_wmma f16,
//          weights streamed ws->LDS via global_load_async_to_lds_b128
//   scatter: dv += gates*eq[src] + scale*dir ; ds += m[:,128:192] (f32 atomics)
//   out = (eq + dv, inv + ds)
// ---------------------------------------------------------------------------

typedef _Float16 half_t;
typedef __attribute__((ext_vector_type(16))) _Float16 v16h;
typedef __attribute__((ext_vector_type(8)))  float    v8f;

#define FDIM     64
#define TE       64      // edges per block
#define NTHREADS 256     // 8 waves (wave32)
#define XA_STR   68      // f16 activation row stride (halves)
#define WT_STR   72      // f16 transposed-weight row stride (144B, 16B-mult.)
#define MP_STR   196     // f16 m-buffer row stride (192 cols + pad)
#define LN_EPS   1e-5f
#define PE_SCALE 0.31415926535897932f   // pi / LENGTH_SCALE

// packed-weight segment offsets in d_ws (halves); all 16B aligned
#define PHI_W1 0
#define PHI_W2 4096
#define PHI_W3 8192      // 192*64
#define W_W1   20480
#define W_W2   24576
#define W_W3   28672
#define WS_HALVES 40960  // total packed halves (81920 bytes)

struct MlpParams {
    const float *w1, *b1, *g1, *be1;
    const float *w2, *b2, *g2, *be2;
    const float *w3, *b3;
};

struct __align__(16) Smem {
    half_t xa[TE][XA_STR];                 //  8,704 B  activations (f16)
    union {
        half_t wt[3 * FDIM][WT_STR];       // 27,648 B  transposed weights (f16)
        float  h[TE][XA_STR];              // 17,408 B  pre-LN hidden (f32)
    } b;
    half_t mp[TE][MP_STR];                 // 25,088 B  m buffer (f16)
    int    srcIdx[TE];
    int    dstIdx[TE];
    float  dist[TE];
    float  dir[TE][3];
};                                          // ~63 KB total

// ---------------------------------------------------------------------------
// Async copy of packed f16 weights (rows x 64 halves, contiguous in ws) into
// LDS rows of stride WT_STR, using GLOBAL_LOAD_ASYNC_TO_LDS_B128 (ASYNCcnt).
// ---------------------------------------------------------------------------
__device__ __forceinline__ void load_weights_async(Smem& s, int tid,
                                                   const half_t* g, int rows) {
    unsigned ldsBase = (unsigned)(unsigned long long)(&s.b.wt[0][0]);
    for (int u = tid; u < rows * 8; u += NTHREADS) {
        int row = u >> 3, ch = u & 7;                    // 8x16B chunks per row
        unsigned laddr = ldsBase + (unsigned)(row * (WT_STR * 2) + ch * 16);
        int goff = row * 128 + ch * 16;                  // bytes into segment
        asm volatile("global_load_async_to_lds_b128 %0, %1, %2"
                     :: "v"(laddr), "v"(goff), "s"(g)
                     : "memory");
    }
    asm volatile("s_wait_asynccnt 0" ::: "memory");
}

// ---------------------------------------------------------------------------
// One 16x16 f32 tile of XA(64x64 f16) @ WT^T via 2x v_wmma_f32_16x16x32_f16.
// Fragment packing used here (wave32; contiguous k order, same for A and B):
//   A 16x32 f16 : M=lane&15; halves[h] = A[M][k0 + 16*(lane>>4) + h]
//   B 32x16 f16 : N=lane&15; halves[h] = B[k0 + 16*(lane>>4) + h][N]
//   C/D f32     : vgpr r -> (M = r + 8*(lane>>4), N = lane&15)
// ---------------------------------------------------------------------------
__device__ __forceinline__ v8f gemm_tile(const half_t* XA, const half_t* WT,
                                         int mt, int nt, int lane) {
    v8f acc = {};
    const int mr = mt * 16 + (lane & 15);
    const int nc = nt * 16 + (lane & 15);
    const int hi = (lane >> 4) & 1;
#pragma unroll
    for (int k0 = 0; k0 < FDIM; k0 += 32) {
        v16h a, b;
        const half_t* arow = XA + mr * XA_STR + k0 + hi * 16;
#pragma unroll
        for (int i = 0; i < 8; ++i) {
            a[i]     = arow[i];        // K = k0 + hi*16 + i
            a[i + 8] = arow[8 + i];    // K = k0 + hi*16 + 8 + i
        }
        const half_t* brow = WT + nc * WT_STR + k0 + hi * 16;
#pragma unroll
        for (int i = 0; i < 16; ++i) b[i] = brow[i];
        acc = __builtin_amdgcn_wmma_f32_16x16x32_f16(
            /*neg_a=*/false, a, /*neg_b=*/false, b,
            /*c_mod=*/(short)0, acc, /*reuse_a=*/false, /*reuse_b=*/false);
    }
    return acc;
}

// Hidden layer: xa(f16) -> GEMM(+bias) -> LayerNorm -> SiLU -> xa(f16)
__device__ void hidden_layer(Smem& s, int tid, const half_t* wtp,
                             const float* B, const float* G, const float* Be) {
    load_weights_async(s, tid, wtp, FDIM);
    __syncthreads();

    const int wave = tid >> 5, lane = tid & 31;
    v8f acc[2];
    int mts[2], nts[2];
#pragma unroll
    for (int t = 0; t < 2; ++t) {
        int tile = wave * 2 + t;          // 16 tiles over 8 waves
        mts[t] = tile >> 2; nts[t] = tile & 3;
        acc[t] = gemm_tile(&s.xa[0][0], &s.b.wt[0][0], mts[t], nts[t], lane);
    }
    __syncthreads();                      // all waves done reading wt

    // store H (f32, +bias) over the weight region
#pragma unroll
    for (int t = 0; t < 2; ++t) {
        int m0 = mts[t] * 16 + ((lane >> 4) & 1) * 8;
        int n  = nts[t] * 16 + (lane & 15);
        float bias = B[n];
#pragma unroll
        for (int r = 0; r < 8; ++r) s.b.h[m0 + r][n] = acc[t][r] + bias;
    }
    __syncthreads();

    // LayerNorm + SiLU: 4 lanes per row, shfl_xor reductions (all waves busy)
    {
        int row = wave * 8 + (lane >> 2);     // 8 waves * 8 rows = 64
        int sub = lane & 3;                   // 16 features per lane
        const float* r = s.b.h[row];
        float sum = 0.f;
#pragma unroll
        for (int i = 0; i < 16; ++i) sum += r[sub * 16 + i];
        sum += __shfl_xor(sum, 1, 32);
        sum += __shfl_xor(sum, 2, 32);
        float mean = sum * (1.0f / FDIM);
        float var = 0.f;
#pragma unroll
        for (int i = 0; i < 16; ++i) {
            float d = r[sub * 16 + i] - mean;
            var += d * d;
        }
        var += __shfl_xor(var, 1, 32);
        var += __shfl_xor(var, 2, 32);
        float rstd = rsqrtf(var * (1.0f / FDIM) + LN_EPS);
#pragma unroll
        for (int i = 0; i < 16; ++i) {
            int f = sub * 16 + i;
            float x = (r[f] - mean) * rstd * G[f] + Be[f];
            float y = x / (1.0f + __expf(-x));   // SiLU
            s.xa[row][f] = (half_t)y;
        }
    }
    __syncthreads();
}

// Output layer: xa(f16) @ W3(64x192) + b3; MUL=false writes mp,
// MUL=true multiplies mp in place (straight-line, no exec branches).
template <bool MUL>
__device__ void output_layer(Smem& s, int tid, const half_t* wtp, const float* B) {
    load_weights_async(s, tid, wtp, 3 * FDIM);
    __syncthreads();

    const int wave = tid >> 5, lane = tid & 31;
#pragma unroll
    for (int t = 0; t < 6; ++t) {
        int tile = wave * 6 + t;          // 48 tiles over 8 waves
        int mt = tile / 12, nt = tile % 12;
        v8f acc = gemm_tile(&s.xa[0][0], &s.b.wt[0][0], mt, nt, lane);
        int m0 = mt * 16 + ((lane >> 4) & 1) * 8;
        int n  = nt * 16 + (lane & 15);
        float bias = B[n];
#pragma unroll
        for (int r = 0; r < 8; ++r) {
            float v = acc[r] + bias;
            if (MUL) v *= (float)s.mp[m0 + r][n];
            s.mp[m0 + r][n] = (half_t)v;
        }
    }
    __syncthreads();
}

__global__ __launch_bounds__(NTHREADS)
void edge_message_kernel(const float* __restrict__ inv,
                         const float* __restrict__ eq,
                         const int*   __restrict__ edge_src,
                         const int*   __restrict__ edge_dst,
                         const float* __restrict__ dist,
                         const float* __restrict__ dirv,
                         const half_t* __restrict__ wts,   // packed f16 weights
                         MlpParams phi, MlpParams wpar,
                         float* out_eq, float* out_inv, int nEdges) {
    __shared__ Smem s;
    const int tid = threadIdx.x;
    const int e0  = blockIdx.x * TE;

    // edge metadata
    if (tid < TE) {
        int e = e0 + tid;
        int ee = (e < nEdges) ? e : (nEdges - 1);
        s.srcIdx[tid] = edge_src[ee];
        s.dstIdx[tid] = edge_dst[ee];
        s.dist[tid]   = dist[ee];
        s.dir[tid][0] = dirv[ee * 3 + 0];
        s.dir[tid][1] = dirv[ee * 3 + 1];
        s.dir[tid][2] = dirv[ee * 3 + 2];
    }
    // warm L2 with the second MLP's packed weights (global_prefetch_b8)
    for (int off = W_W1 * 2 + tid * 128; off < WS_HALVES * 2; off += NTHREADS * 128)
        __builtin_prefetch((const char*)wts + off, 0, 1);
    __syncthreads();

    // gather inv[src] -> xa (f16)
    for (int idx = tid; idx < TE * FDIM; idx += NTHREADS) {
        int e = idx >> 6, f = idx & 63;
        s.xa[e][f] = (half_t)inv[s.srcIdx[e] * FDIM + f];
    }
    __syncthreads();

    // phi MLP
    hidden_layer(s, tid, wts + PHI_W1, phi.b1, phi.g1, phi.be1);
    hidden_layer(s, tid, wts + PHI_W2, phi.b2, phi.g2, phi.be2);
    output_layer<false>(s, tid, wts + PHI_W3, phi.b3);

    // positional encoding -> xa (interleaved cos/sin, ranks 1..32)
    for (int idx = tid; idx < TE * FDIM; idx += NTHREADS) {
        int e = idx >> 6, c = idx & 63;
        float rank = (float)((c >> 1) + 1);
        float ang  = s.dist[e] * PE_SCALE * rank;
        float v    = (c & 1) ? __sinf(ang) : __cosf(ang);
        s.xa[e][c] = (half_t)v;
    }
    __syncthreads();

    // w MLP, fused elementwise product into mp
    hidden_layer(s, tid, wts + W_W1, wpar.b1, wpar.g1, wpar.be1);
    hidden_layer(s, tid, wts + W_W2, wpar.b2, wpar.g2, wpar.be2);
    output_layer<true>(s, tid, wts + W_W3, wpar.b3);

    // fused scatter: dv += gates*eq[src] + scale*dir ; ds += m[:,128:192]
    for (int idx = tid; idx < TE * FDIM; idx += NTHREADS) {
        int e = idx >> 6, c = idx & 63;
        if (e0 + e < nEdges) {
            float g  = (float)s.mp[e][c];
            float sc = (float)s.mp[e][c + 64];
            float d  = (float)s.mp[e][c + 128];
            int sn = s.srcIdx[e], dn = s.dstIdx[e];
#pragma unroll
            for (int k = 0; k < 3; ++k) {
                float ev = eq[(sn * FDIM + c) * 3 + k];
                atomicAdd(&out_eq[(dn * FDIM + c) * 3 + k],
                          g * ev + sc * s.dir[e][k]);
            }
            atomicAdd(&out_inv[dn * FDIM + c], d);
        }
    }
}

// Pack all 6 weight matrices: f32 [fi][fo] -> f16 transposed [fo][fi] in ws
__global__ void pack_weights_kernel(MlpParams phi, MlpParams wpar,
                                    half_t* __restrict__ out) {
    int i = blockIdx.x * blockDim.x + threadIdx.x;
    if (i >= WS_HALVES) return;
    const float* src; int cols; int base;
    if      (i < PHI_W2) { src = phi.w1;  cols = 64;  base = PHI_W1; }
    else if (i < PHI_W3) { src = phi.w2;  cols = 64;  base = PHI_W2; }
    else if (i < W_W1)   { src = phi.w3;  cols = 192; base = PHI_W3; }
    else if (i < W_W2)   { src = wpar.w1; cols = 64;  base = W_W1; }
    else if (i < W_W3)   { src = wpar.w2; cols = 64;  base = W_W2; }
    else                 { src = wpar.w3; cols = 192; base = W_W3; }
    int j = i - base;                 // j = n*64 + k (transposed layout)
    int n = j >> 6, k = j & 63;
    out[i] = (half_t)src[k * cols + n];
}

// out = concat(eq, inv) before the atomic scatter accumulates into it
__global__ void init_out_kernel(const float* __restrict__ eq,
                                const float* __restrict__ inv,
                                float* __restrict__ out, int nEq, int nInv) {
    int i = blockIdx.x * blockDim.x + threadIdx.x;
    int total = nEq + nInv;
    if (i < total) out[i] = (i < nEq) ? eq[i] : inv[i - nEq];
}

extern "C" void kernel_launch(void* const* d_in, const int* in_sizes, int n_in,
                              void* d_out, int out_size, void* d_ws, size_t ws_size,
                              hipStream_t stream) {
    const float* inv  = (const float*)d_in[0];
    const float* eq   = (const float*)d_in[1];
    const int*   eidx = (const int*)d_in[2];
    const float* dist = (const float*)d_in[3];
    const float* dirv = (const float*)d_in[4];

    MlpParams phi, wpar;
    const float** pf = (const float**)&phi;
    const float** wf = (const float**)&wpar;
    for (int i = 0; i < 10; ++i) {
        pf[i] = (const float*)d_in[5 + i];       // w1,b1,g1,be1,w2,b2,g2,be2,w3,b3
        wf[i] = (const float*)d_in[15 + i];
    }

    const int E    = in_sizes[3];                // edge_dist length
    const int nEq  = in_sizes[1];                // N*64*3
    const int nInv = in_sizes[0];                // N*64
    float* out_eq  = (float*)d_out;
    float* out_inv = out_eq + nEq;
    half_t* wts    = (half_t*)d_ws;

    pack_weights_kernel<<<(WS_HALVES + 255) / 256, 256, 0, stream>>>(phi, wpar, wts);

    int total = nEq + nInv;
    init_out_kernel<<<(total + 255) / 256, 256, 0, stream>>>(eq, inv, out_eq, nEq, nInv);

    int nBlocks = (E + TE - 1) / TE;
    edge_message_kernel<<<nBlocks, NTHREADS, 0, stream>>>(
        inv, eq, eidx, eidx + E, dist, dirv, wts, phi, wpar, out_eq, out_inv, E);
}
